// PAM_Module_8297876816356
// MI455X (gfx1250) — compile-verified, matrix operand order checked by reading, not run on hardware
//
#include <hip/hip_runtime.h>

// Problem constants
#define B_   2
#define C_   512
#define N_   4096          // H*W
#define BN_  8192          // B*N

typedef __attribute__((ext_vector_type(16))) _Float16 v16h;
typedef __attribute__((ext_vector_type(8)))  _Float16 v8h;
typedef __attribute__((ext_vector_type(8)))  float    v8f;
typedef __attribute__((ext_vector_type(4)))  unsigned int u32x4;
typedef __attribute__((ext_vector_type(8)))  int      i32x8;
typedef __attribute__((ext_vector_type(4)))  int      i32x4;

union F16Frag { v16h v; v8h h[2]; };

// 16x32 f16 A/B fragment: lane<16 holds K={0..7,16..23}, lane>=16 K={8..15,24..31}.
// k0 = c_base + (lane>>4)*8.
static __device__ __forceinline__ v16h make_frag(const _Float16* base, int k0) {
    F16Frag f;
    f.h[0] = *(const v8h*)(base + k0);
    f.h[1] = *(const v8h*)(base + k0 + 16);
    return f.v;
}

static __device__ __forceinline__ v8f wmma_f16(v16h a, v16h b, v8f c) {
    return __builtin_amdgcn_wmma_f32_16x16x32_f16(false, a, false, b, (short)0, c, false, false);
}

// ---------------- TDM staging (CDNA5 Tensor Data Mover) ----------------

#if __has_builtin(__builtin_amdgcn_tensor_load_to_lds) && __has_builtin(__builtin_amdgcn_s_wait_tensorcnt)
#define TDM_OK 1
#else
#define TDM_OK 0
#endif

#if TDM_OK
// Issue one 2D tensor-tile DMA: global (row-major, stride0 elements between
// rows, 2-byte elements) -> LDS (packed).  Descriptor per ISA 08_async_tensor
// sec 8.3-8.5: group0 = {count/type, lds_addr, global_addr57, type=2},
// group1 = {data_size, tensor_dim0/1, tile_dim0/1, tensor_dim0_stride}.
static __device__ __forceinline__ void tdm_load_2d(
        unsigned lds_off, const void* gptr,
        unsigned tensor_d0, unsigned tensor_d1,
        unsigned tile_d0, unsigned tile_d1,
        unsigned long long stride0) {
    unsigned long long ga = (unsigned long long)(size_t)gptr;
    u32x4 g0;
    g0[0] = 1u;                                            // count=1, user desc
    g0[1] = lds_off;                                       // LDS byte address
    g0[2] = (unsigned)(ga & 0xffffffffull);                // global_addr[31:0]
    g0[3] = (unsigned)((ga >> 32) & 0x01ffffffull)         // global_addr[56:32]
          | (2u << 30);                                    // type = 2 (image)
    i32x8 g1;
    g1[0] = (int)(1u << 16);                               // data_size=1 -> 2B
    g1[1] = (int)((tensor_d0 & 0xffffu) << 16);            // td0[15:0] @ [63:48]
    g1[2] = (int)((tensor_d0 >> 16) | ((tensor_d1 & 0xffffu) << 16));
    g1[3] = (int)((tensor_d1 >> 16) | (tile_d0 << 16));    // tile_dim0 @ [127:112]
    g1[4] = (int)tile_d1;                                  // tile_dim1, tile_dim2=0
    g1[5] = (int)(stride0 & 0xffffffffull);                // dim0_stride low
    g1[6] = (int)((stride0 >> 32) & 0xffffull);            // dim0_stride high
    g1[7] = 0;                                             // dim1_stride unused (2D)
    i32x4 z4 = {0, 0, 0, 0};
#if defined(__clang_major__) && (__clang_major__ >= 23)
    i32x8 z8 = {0, 0, 0, 0, 0, 0, 0, 0};
    __builtin_amdgcn_tensor_load_to_lds(g0, g1, z4, z4, z8, 0);
#else
    __builtin_amdgcn_tensor_load_to_lds(g0, g1, z4, z4, 0);
#endif
}
#endif

// ---------------- phase 0: dtype conversion ----------------

__global__ void pam_cvt_x(const float* __restrict__ x, _Float16* __restrict__ xh, int n) {
    int i = blockIdx.x * blockDim.x + threadIdx.x;
    if (i < n) xh[i] = (_Float16)x[i];
}

// wt[w][co][ci] = W_w[ci][co] in f16 (transposed so B fragments load contiguously)
__global__ void pam_cvt_w(const float* __restrict__ Wq, const float* __restrict__ Wk,
                          const float* __restrict__ Wv, _Float16* __restrict__ wt) {
    int i = blockIdx.x * blockDim.x + threadIdx.x;
    if (i >= 3 * C_ * C_) return;
    int w  = i / (C_ * C_);
    int r  = i % (C_ * C_);
    int co = r / C_;
    int ci = r % C_;
    const float* Ws = (w == 0) ? Wq : (w == 1) ? Wk : Wv;
    wt[i] = (_Float16)Ws[ci * C_ + co];
}

// ---------------- phase 1: QKV projection GEMM ----------------
// One wave per 16x16 output tile.  which = 0:Q row-major, 1:K row-major, 2:V transposed.

__global__ __launch_bounds__(32)
void pam_qkv(const _Float16* __restrict__ xh, const _Float16* __restrict__ wt,
             const float* __restrict__ bq, const float* __restrict__ bk,
             const float* __restrict__ bv,
             _Float16* __restrict__ q, _Float16* __restrict__ k,
             _Float16* __restrict__ vt) {
    const int lane = threadIdx.x;
    const int r    = lane & 15;
    const int hf   = lane >> 4;
    const int n0   = blockIdx.x * 16;       // global row (across batches)
    const int co0  = blockIdx.y * 16;
    const int which = blockIdx.z;

    const _Float16* arow = xh + (size_t)(n0 + r) * C_;
    const _Float16* brow = wt + (size_t)which * C_ * C_ + (size_t)(co0 + r) * C_;

    v8f acc = {};
    #pragma unroll
    for (int kk = 0; kk < C_; kk += 32) {
        v16h a = make_frag(arow, kk + hf * 8);
        v16h b = make_frag(brow, kk + hf * 8);
        acc = wmma_f16(a, b, acc);
    }

    const float* bias = (which == 0) ? bq : (which == 1) ? bk : bv;
    const float bc = bias[co0 + r];
    #pragma unroll
    for (int j = 0; j < 8; ++j) acc[j] += bc;

    if (which == 2) {
        // transposed store: lane holds column co0+r, rows hf*8+j -> one b128
        v8h pk;
        #pragma unroll
        for (int j = 0; j < 8; ++j) pk[j] = (_Float16)acc[j];
        const int bb = n0 >> 12;
        const int nl = (n0 & (N_ - 1)) + hf * 8;
        _Float16* dst = vt + ((size_t)bb * C_ + (co0 + r)) * (size_t)N_ + nl;
        *(v8h*)dst = pk;
    } else {
        __shared__ __align__(16) _Float16 sm[16][16];
        #pragma unroll
        for (int j = 0; j < 8; ++j) sm[j + hf * 8][r] = (_Float16)acc[j];
        __syncthreads();
        const int rr = lane >> 1, hh = lane & 1;
        v8h val = *(const v8h*)&sm[rr][hh * 8];
        _Float16* dst = ((which == 0) ? q : k) + (size_t)(n0 + rr) * C_ + co0 + hh * 8;
        *(v8h*)dst = val;
    }
}

// ---------------- phase 2: streaming (flash) attention ----------------
// Block = 8 waves over a 64-row n-block; wave w: n-tile = w>>1, c-slice = (w&1)*256.
// K (64x512), Q (32x512) and Vt (512x32) tiles live in LDS, staged by the TDM.
// energy[n,m] = sum_c K[n,c] Q[m,c]; softmax over m; out = P @ V.

#define SK_OFF    0                 // 64*512 f16  = 64 KB
#define SQ_OFF    65536             // 32*512 f16  = 32 KB
#define SV_OFF    98304             // 512*32 f16  = 32 KB
#define PB_OFF    131072            // 8 waves * 16*32 f16 = 8 KB
#define LDS_BYTES 139264

__global__ __launch_bounds__(256)
void pam_attn(const _Float16* __restrict__ q, const _Float16* __restrict__ k,
              const _Float16* __restrict__ vt, const float* __restrict__ x,
              const float* __restrict__ betap, float* __restrict__ out) {
    extern __shared__ char smem[];
    _Float16* sK = (_Float16*)(smem + SK_OFF);   // [64][512]
    _Float16* sQ = (_Float16*)(smem + SQ_OFF);   // [32][512]
    _Float16* sV = (_Float16*)(smem + SV_OFF);   // [512][32]

    const int tid  = threadIdx.x;
    const int lane = tid & 31;
    const int wv   = tid >> 5;          // 0..7
    const int r    = lane & 15;
    const int hf   = lane >> 4;
    const int nt   = wv >> 1;           // n-tile within the 64-row block
    const int ch0  = (wv & 1) * 256;    // c-slice
    const int bb   = blockIdx.y;
    const int n0   = blockIdx.x * 64;   // within-batch row base

    const _Float16* qb = q  + (size_t)bb * N_ * C_;
    const _Float16* kb = k  + (size_t)bb * N_ * C_;
    const _Float16* vb = vt + (size_t)bb * C_ * N_;
    _Float16* pb = (_Float16*)(smem + PB_OFF) + wv * (16 * 32);

    // ---- prologue staging: K block + first Q/V tiles ----
#if TDM_OK
    if (tid < 32) {
        tdm_load_2d((unsigned)(size_t)(void*)sK, kb + (size_t)n0 * C_,
                    C_, N_, C_, 64, C_);
        tdm_load_2d((unsigned)(size_t)(void*)sQ, qb,
                    C_, N_, C_, 32, C_);
        tdm_load_2d((unsigned)(size_t)(void*)sV, vb,
                    N_, C_, 32, C_, N_);
        __builtin_amdgcn_s_wait_tensorcnt(0);
    }
#else
    for (int i = tid * 8; i < 64 * C_; i += 256 * 8)
        *(v8h*)(sK + i) = *(const v8h*)(kb + (size_t)n0 * C_ + i);
    for (int i = tid * 8; i < 32 * C_; i += 256 * 8)
        *(v8h*)(sQ + i) = *(const v8h*)(qb + i);
    for (int idx = tid; idx < C_ * 4; idx += 256) {
        int row = idx >> 2, off = (idx & 3) * 8;
        *(v8h*)(sV + row * 32 + off) = *(const v8h*)(vb + (size_t)row * N_ + off);
    }
#endif
    __syncthreads();

    v8f z = {};
    v8f oacc[16];
    #pragma unroll
    for (int cb = 0; cb < 16; ++cb) oacc[cb] = z;

    float rm[8], rl[8];
    #pragma unroll
    for (int j = 0; j < 8; ++j) { rm[j] = -__builtin_inff(); rl[j] = 0.0f; }

    const _Float16* krow = sK + (size_t)(nt * 16 + r) * C_;

    for (int m0 = 0; m0 < N_; m0 += 32) {
        const _Float16* q0row = sQ + (size_t)r * C_;
        const _Float16* q1row = sQ + (size_t)(16 + r) * C_;

        // S tiles: rows = n (this wave's 16 K rows), cols = m
        v8f s0 = {}, s1 = {};
        #pragma unroll
        for (int kk = 0; kk < C_; kk += 32) {
            v16h ka = make_frag(krow,  kk + hf * 8);
            v16h qa = make_frag(q0row, kk + hf * 8);
            v16h qc = make_frag(q1row, kk + hf * 8);
            s0 = wmma_f16(ka, qa, s0);
            s1 = wmma_f16(ka, qc, s1);
        }

        // online softmax over 32 new columns; row j+8*hf lives across 16 lanes
        float p0[8], p1[8], al[8];
        #pragma unroll
        for (int j = 0; j < 8; ++j) {
            float mx = fmaxf(s0[j], s1[j]);
            mx = fmaxf(mx, __shfl_xor(mx, 1));
            mx = fmaxf(mx, __shfl_xor(mx, 2));
            mx = fmaxf(mx, __shfl_xor(mx, 4));
            mx = fmaxf(mx, __shfl_xor(mx, 8));
            float nm = fmaxf(rm[j], mx);
            al[j] = __expf(rm[j] - nm);
            p0[j] = __expf(s0[j] - nm);
            p1[j] = __expf(s1[j] - nm);
            float rs = p0[j] + p1[j];
            rs += __shfl_xor(rs, 1);
            rs += __shfl_xor(rs, 2);
            rs += __shfl_xor(rs, 4);
            rs += __shfl_xor(rs, 8);
            rl[j] = rl[j] * al[j] + rs;
            rm[j] = nm;
        }
        #pragma unroll
        for (int cb = 0; cb < 16; ++cb)
            #pragma unroll
            for (int j = 0; j < 8; ++j) oacc[cb][j] *= al[j];

        // transpose P (acc layout) -> A-fragment layout via per-wave LDS tile
        #pragma unroll
        for (int j = 0; j < 8; ++j) {
            pb[(j + hf * 8) * 32 + r]      = (_Float16)p0[j];
            pb[(j + hf * 8) * 32 + 16 + r] = (_Float16)p1[j];
        }
        F16Frag pf;
        pf.h[0] = *(const v8h*)(pb + r * 32 + hf * 8);
        pf.h[1] = *(const v8h*)(pb + r * 32 + hf * 8 + 16);

        // out += P(16x32) @ V(32x16) per c-block; sV rows give contiguous chunks
        #pragma unroll
        for (int cb = 0; cb < 16; ++cb) {
            const _Float16* vrow = sV + (ch0 + cb * 16 + r) * 32;
            F16Frag vf;
            vf.h[0] = *(const v8h*)(vrow + hf * 8);
            vf.h[1] = *(const v8h*)(vrow + hf * 8 + 16);
            oacc[cb] = wmma_f16(pf.v, vf.v, oacc[cb]);
        }

        __syncthreads();                      // all waves done with sQ/sV
        if (m0 + 32 < N_) {
            const int mn = m0 + 32;
#if TDM_OK
            if (tid < 32) {
                tdm_load_2d((unsigned)(size_t)(void*)sQ, qb + (size_t)mn * C_,
                            C_, N_, C_, 32, C_);
                tdm_load_2d((unsigned)(size_t)(void*)sV, vb + mn,
                            N_, C_, 32, C_, N_);
                __builtin_amdgcn_s_wait_tensorcnt(0);
            }
#else
            for (int i = tid * 8; i < 32 * C_; i += 256 * 8)
                *(v8h*)(sQ + i) = *(const v8h*)(qb + (size_t)mn * C_ + i);
            for (int idx = tid; idx < C_ * 4; idx += 256) {
                int row = idx >> 2, off = (idx & 3) * 8;
                *(v8h*)(sV + row * 32 + off) =
                    *(const v8h*)(vb + (size_t)row * N_ + mn + off);
            }
#endif
        }
        __syncthreads();                      // next tiles visible to all
    }

    const float beta = betap[0];
    #pragma unroll
    for (int j = 0; j < 8; ++j) rl[j] = 1.0f / rl[j];   // row sums >= 1, safe
    #pragma unroll
    for (int cb = 0; cb < 16; ++cb) {
        const int c = ch0 + cb * 16 + r;
        #pragma unroll
        for (int j = 0; j < 8; ++j) {
            const size_t idx = ((size_t)(bb * N_ + n0 + nt * 16 + hf * 8 + j)) * C_ + c;
            out[idx] = beta * (oacc[cb][j] * rl[j]) + x[idx];
        }
    }
}

// ---------------- launcher ----------------

extern "C" void kernel_launch(void* const* d_in, const int* in_sizes, int n_in,
                              void* d_out, int out_size, void* d_ws, size_t ws_size,
                              hipStream_t stream) {
    const float* x    = (const float*)d_in[0];
    const float* Wq   = (const float*)d_in[1];
    const float* bq   = (const float*)d_in[2];
    const float* Wk   = (const float*)d_in[3];
    const float* bk   = (const float*)d_in[4];
    const float* Wv   = (const float*)d_in[5];
    const float* bv   = (const float*)d_in[6];
    const float* beta = (const float*)d_in[7];
    float* out = (float*)d_out;

    // workspace layout (f16): xh | wt(3 transposed) | q | k | vt
    char* w = (char*)d_ws;
    const size_t XH = (size_t)BN_ * C_ * 2;        // 8 MB
    const size_t WT = (size_t)3 * C_ * C_ * 2;     // 1.5 MB
    const size_t QK = (size_t)BN_ * C_ * 2;        // 8 MB each
    _Float16* xh = (_Float16*)(w);
    _Float16* wt = (_Float16*)(w + XH);
    _Float16* qh = (_Float16*)(w + XH + WT);
    _Float16* kh = (_Float16*)(w + XH + WT + QK);
    _Float16* vt = (_Float16*)(w + XH + WT + 2 * QK);

    pam_cvt_x<<<(BN_ * C_) / 256, 256, 0, stream>>>(x, xh, BN_ * C_);
    pam_cvt_w<<<(3 * C_ * C_) / 256, 256, 0, stream>>>(Wq, Wk, Wv, wt);

    pam_qkv<<<dim3(BN_ / 16, C_ / 16, 3), 32, 0, stream>>>(
        xh, wt, bq, bk, bv, qh, kh, vt);

    pam_attn<<<dim3(N_ / 64, B_, 1), 256, LDS_BYTES, stream>>>(
        qh, kh, vt, x, beta, out);
}
